// CrossAttention_61048665145417
// MI455X (gfx1250) — matrix operand order repaired and checked by reading, not compiled, not run on hardware
//
#include <hip/hip_runtime.h>
#include <type_traits>
#include <cstddef>

// ---------------------------------------------------------------------------
// CDNA5 (gfx1250) cross-attention, WMMA f32_16x16x32_f16 throughout.
// ---------------------------------------------------------------------------

typedef _Float16 half_t;
typedef __attribute__((ext_vector_type(16))) _Float16 v16h;
typedef __attribute__((ext_vector_type(8)))  _Float16 v8h;
typedef __attribute__((ext_vector_type(8)))  float    v8f;
typedef __attribute__((ext_vector_type(4)))  float    v4f;

#define B_     16
#define LQ     4096
#define LKV    77
#define KVPAD  80     // kv padded to 5 full 16-wide score tiles (branchless K loop)
#define DEMB   1024
#define DCROSS 768
#define KVP    96     // kv padded to 3 K-tiles of 32 for the attn@V WMMA

#define WMMA_F16(a, b, c) \
  __builtin_amdgcn_wmma_f32_16x16x32_f16(false, (a), false, (b), (short)0, (c), false, false)

__device__ inline v8f zero_v8f() {
  v8f z;
#pragma unroll
  for (int i = 0; i < 8; ++i) z[i] = 0.0f;
  return z;
}

// A-fragment load for V_WMMA_F32_16X16X32_F16 (16x32 MxK, row-major source).
// lanes 0-15: K+0..15; lanes 16-31: K+16..31 (contiguous, matches B).
template <typename AT>
__device__ inline v16h load_a_frag(const AT* __restrict__ A, int K, int row, int k0, int lane) {
  const int sel = (lane & 16) ? 16 : 0;
  if constexpr (std::is_same<AT, float>::value) {
    const float* p = A + (size_t)row * K + k0 + sel;
    v4f f0 = *(const v4f*)(p);
    v4f f1 = *(const v4f*)(p + 4);
    v4f f2 = *(const v4f*)(p + 8);
    v4f f3 = *(const v4f*)(p + 12);
    v16h a;
#pragma unroll
    for (int i = 0; i < 4; ++i) {
      a[i]      = (_Float16)f0[i];
      a[i + 4]  = (_Float16)f1[i];
      a[i + 8]  = (_Float16)f2[i];
      a[i + 12] = (_Float16)f3[i];
    }
    return a;
  } else {
    const half_t* p = A + (size_t)row * K + k0 + sel;
    v8h lo = *(const v8h*)(p);
    v8h hi = *(const v8h*)(p + 8);
    return __builtin_shufflevector(lo, hi, 0, 1, 2, 3, 4, 5, 6, 7,
                                   8, 9, 10, 11, 12, 13, 14, 15);
  }
}

// ---------------------------------------------------------------------------
// Weight transpose + f32->f16 convert:  W[K][N] (row-major f32) -> Wt[N][K] f16
// ---------------------------------------------------------------------------
__global__ __launch_bounds__(256)
void transpose_convert_kernel(const float* __restrict__ W, half_t* __restrict__ Wt,
                              int K, int N) {
  __shared__ float tile[32][33];
  const int tx = threadIdx.x;   // 0..31
  const int ty = threadIdx.y;   // 0..7
  const int n0 = blockIdx.x * 32;
  const int k0 = blockIdx.y * 32;
  for (int r = ty; r < 32; r += 8)
    tile[r][tx] = W[(size_t)(k0 + r) * N + (n0 + tx)];
  __syncthreads();
  for (int r = ty; r < 32; r += 8)
    Wt[(size_t)(n0 + r) * K + (k0 + tx)] = (half_t)tile[tx][r];
}

// ---------------------------------------------------------------------------
// GEMM: C[M][N] = A[M][K] * Bt[N][K]^T + bias[N]
// Block = 128 threads (4 waves). Wave tile = (MT*16) x 64; waves cover
// distinct M strips, so block tile = (4*MT*16) x 64.
// Requirements: N % 64 == 0, K % 32 == 0, M % 16 == 0 (guarded per wave).
// ---------------------------------------------------------------------------
template <int MT, typename AT, typename CT>
__global__ __launch_bounds__(128)
void gemm_bt_kernel(const AT* __restrict__ A, const half_t* __restrict__ Bt,
                    const float* __restrict__ bias, CT* __restrict__ C,
                    int M, int N, int K) {
  const int lane  = threadIdx.x & 31;
  const int wave  = threadIdx.x >> 5;
  const int m0    = blockIdx.y * (4 * MT * 16) + wave * (MT * 16);
  const int n0    = blockIdx.x * 64;
  if (m0 >= M) return;                       // wave-level tail guard (no barriers used)
  const int cwl   = lane & 15;
  const int sel16 = (lane & 16) ? 16 : 0;

  v8f acc[MT][4];
#pragma unroll
  for (int mt = 0; mt < MT; ++mt)
#pragma unroll
    for (int t = 0; t < 4; ++t) acc[mt][t] = zero_v8f();

  for (int k0 = 0; k0 < K; k0 += 32) {
    v16h a[MT];
#pragma unroll
    for (int mt = 0; mt < MT; ++mt)
      a[mt] = load_a_frag<AT>(A, K, m0 + mt * 16 + cwl, k0, lane);
    const half_t* bbase = Bt + (size_t)(n0 + cwl) * K + k0 + sel16;
#pragma unroll
    for (int t = 0; t < 4; ++t) {
      v16h b = *(const v16h*)(bbase + (size_t)t * 16 * K);
#pragma unroll
      for (int mt = 0; mt < MT; ++mt)
        acc[mt][t] = WMMA_F16(a[mt], b, acc[mt][t]);
    }
  }

#pragma unroll
  for (int mt = 0; mt < MT; ++mt) {
    const int rbase = m0 + mt * 16 + ((lane & 16) ? 8 : 0);
#pragma unroll
    for (int t = 0; t < 4; ++t) {
      const int col = n0 + t * 16 + cwl;
      const float bv = bias[col];
#pragma unroll
      for (int j = 0; j < 8; ++j) {
        const float v = acc[mt][t][j] + bv;
        CT* cp = C + (size_t)(rbase + j) * N + col;
        if constexpr (std::is_same<CT, half_t>::value) *cp = (half_t)v;
        else                                           *cp = v;
      }
    }
  }
}

// ---------------------------------------------------------------------------
// Khp[b][kv<KVPAD][e] = (kv < LKV) ? Kh[b][kv][e] : 0   (branchless score loop)
// ---------------------------------------------------------------------------
__global__ __launch_bounds__(256)
void pad_k_kernel(const half_t* __restrict__ Kh, half_t* __restrict__ Khp) {
  const int idx = blockIdx.x * blockDim.x + threadIdx.x;   // over B_*KVPAD*DEMB
  if (idx >= B_ * KVPAD * DEMB) return;
  const int e  = idx % DEMB;
  const int kv = (idx / DEMB) % KVPAD;
  const int b  = idx / (DEMB * KVPAD);
  half_t v = (half_t)0.0f;
  if (kv < LKV) v = Kh[((size_t)b * LKV + kv) * DEMB + e];
  Khp[idx] = v;
}

// ---------------------------------------------------------------------------
// Vt[b][e][kv_pad] = Vh[b][kv][e] (f16, kv padded 77 -> 96 with zeros)
// ---------------------------------------------------------------------------
__global__ __launch_bounds__(256)
void build_vt_kernel(const half_t* __restrict__ Vh, half_t* __restrict__ Vt) {
  const int idx = blockIdx.x * blockDim.x + threadIdx.x;
  if (idx >= B_ * DEMB * KVP) return;
  const int l = idx % KVP;
  const int e = (idx / KVP) % DEMB;
  const int b = idx / (KVP * DEMB);
  half_t v = (half_t)0.0f;
  if (l < LKV) v = Vh[((size_t)b * LKV + l) * DEMB + e];
  Vt[idx] = v;
}

// ---------------------------------------------------------------------------
// Fused attention: scores = Qh Khp^T * scale -> softmax -> ctx = attn * V
// One wave per 16-query block; 8 waves (256 thr) per workgroup.
// ---------------------------------------------------------------------------
__global__ __launch_bounds__(256)
void attention_kernel(const half_t* __restrict__ Qh, const half_t* __restrict__ Khp,
                      const half_t* __restrict__ Vt, half_t* __restrict__ O) {
  __shared__ __align__(16) half_t atile[8][16][KVP];

  const int lane  = threadIdx.x & 31;
  const int wave  = threadIdx.x >> 5;
  const int g     = blockIdx.x * 8 + wave;   // 0 .. B*LQ/16 - 1
  const int b     = g >> 8;                  // / (LQ/16 == 256)
  const int q0    = (g & 255) * 16;
  const int cwl   = lane & 15;
  const int sel8  = (lane & 16) ? 8 : 0;
  const int sel16 = (lane & 16) ? 16 : 0;

  // Zero this wave's attn staging tile (covers kv padding 80..95).
  {
    unsigned int* z = (unsigned int*)&atile[wave][0][0];
    for (int i = lane; i < 16 * KVP / 2; i += 32) z[i] = 0u;
  }

  // ---- Stage 1: scores[16 x 80] over K=1024, branchless -------------------
  v8f acc[5];
#pragma unroll
  for (int t = 0; t < 5; ++t) acc[t] = zero_v8f();

  const half_t* Qrow = Qh + (size_t)(b * LQ + q0 + cwl) * DEMB;
  const half_t* Kcol = Khp + ((size_t)b * KVPAD + cwl) * DEMB;   // kv = cwl + kt*16
  for (int k0 = 0; k0 < DEMB; k0 += 32) {
    const half_t* p = Qrow + k0 + sel16;
    v8h lo = *(const v8h*)(p);
    v8h hi = *(const v8h*)(p + 8);
    v16h a = __builtin_shufflevector(lo, hi, 0, 1, 2, 3, 4, 5, 6, 7,
                                     8, 9, 10, 11, 12, 13, 14, 15);
    const half_t* kb = Kcol + k0 + sel16;
#pragma unroll
    for (int kt = 0; kt < 5; ++kt) {
      v16h bf = *(const v16h*)(kb + (size_t)kt * 16 * DEMB);
      acc[kt] = WMMA_F16(a, bf, acc[kt]);
    }
  }

  // ---- Stage 2: scale, mask, row softmax (rows live across 16-lane halves)
  const float scale = 0.08838834764831845f;   // 1/sqrt(128)
#pragma unroll
  for (int kt = 0; kt < 5; ++kt) {
    const bool ok = (kt * 16 + cwl) < LKV;
#pragma unroll
    for (int j = 0; j < 8; ++j)
      acc[kt][j] = ok ? acc[kt][j] * scale : -__builtin_inff();
  }
#pragma unroll
  for (int j = 0; j < 8; ++j) {
    float m = acc[0][j];
#pragma unroll
    for (int kt = 1; kt < 5; ++kt) m = fmaxf(m, acc[kt][j]);
#pragma unroll
    for (int d = 1; d < 16; d <<= 1) m = fmaxf(m, __shfl_xor(m, d));  // stays in 16-lane half
    float s = 0.0f;
#pragma unroll
    for (int kt = 0; kt < 5; ++kt) {
      const float p = __expf(acc[kt][j] - m);   // exp(-inf)=0 handles mask
      acc[kt][j] = p;
      s += p;
    }
#pragma unroll
    for (int d = 1; d < 16; d <<= 1) s += __shfl_xor(s, d);
    const float inv = 1.0f / s;
#pragma unroll
    for (int kt = 0; kt < 5; ++kt) acc[kt][j] *= inv;
  }

  // Stage C/D-layout attn into row-major LDS tile [16][KVP] as f16.
#pragma unroll
  for (int kt = 0; kt < 5; ++kt)
#pragma unroll
    for (int j = 0; j < 8; ++j)
      atile[wave][sel8 + j][kt * 16 + cwl] = (half_t)acc[kt][j];
  __syncthreads();

  // ---- Stage 3: ctx[16 x 1024] = attn[16 x 96] * V[96 x 1024] -------------
  v16h a2[3];
#pragma unroll
  for (int t = 0; t < 3; ++t) {
    const half_t* p = &atile[wave][cwl][t * 32 + sel16];
    v8h lo = *(const v8h*)(p);
    v8h hi = *(const v8h*)(p + 8);
    a2[t] = __builtin_shufflevector(lo, hi, 0, 1, 2, 3, 4, 5, 6, 7,
                                    8, 9, 10, 11, 12, 13, 14, 15);
  }
  const half_t* VtB = Vt + (size_t)b * DEMB * KVP;
  for (int e0 = 0; e0 < DEMB; e0 += 16) {
    v8f oacc = zero_v8f();
#pragma unroll
    for (int t = 0; t < 3; ++t) {
      v16h bf = *(const v16h*)(VtB + (size_t)(e0 + cwl) * KVP + t * 32 + sel16);
      oacc = WMMA_F16(a2[t], bf, oacc);
    }
    half_t* orow = O + (size_t)(b * LQ + q0 + sel8) * DEMB + e0 + cwl;
#pragma unroll
    for (int j = 0; j < 8; ++j)
      orow[(size_t)j * DEMB] = (half_t)oacc[j];
  }
}

// ---------------------------------------------------------------------------
// Host-side orchestration
// ---------------------------------------------------------------------------
extern "C" void kernel_launch(void* const* d_in, const int* in_sizes, int n_in,
                              void* d_out, int out_size, void* d_ws, size_t ws_size,
                              hipStream_t stream) {
  const float* x  = (const float*)d_in[0];
  const float* y  = (const float*)d_in[1];
  const float* Wq = (const float*)d_in[2];
  const float* bq = (const float*)d_in[3];
  const float* Wk = (const float*)d_in[4];
  const float* bk = (const float*)d_in[5];
  const float* Wv = (const float*)d_in[6];
  const float* bv = (const float*)d_in[7];
  const float* Wo = (const float*)d_in[8];
  const float* bo = (const float*)d_in[9];
  float* out = (float*)d_out;

  // Workspace carve-out (~287 MB total).
  char* w = (char*)d_ws;
  auto take = [&](size_t bytes) {
    char* p = w;
    w += (bytes + 255) & ~(size_t)255;
    return p;
  };
  half_t* Qh  = (half_t*)take((size_t)B_ * LQ * DEMB * 2);     // 134 MB
  half_t* Oh  = (half_t*)take((size_t)B_ * LQ * DEMB * 2);     // 134 MB
  half_t* Kh  = (half_t*)take((size_t)B_ * LKV * DEMB * 2);
  half_t* Khp = (half_t*)take((size_t)B_ * KVPAD * DEMB * 2);
  half_t* Vh  = (half_t*)take((size_t)B_ * LKV * DEMB * 2);
  half_t* Vt  = (half_t*)take((size_t)B_ * DEMB * KVP * 2);
  half_t* WqT = (half_t*)take((size_t)DEMB * DEMB * 2);
  half_t* WkT = (half_t*)take((size_t)DEMB * DCROSS * 2);
  half_t* WvT = (half_t*)take((size_t)DEMB * DCROSS * 2);
  half_t* WoT = (half_t*)take((size_t)DEMB * DEMB * 2);

  const dim3 tb(32, 8);
  transpose_convert_kernel<<<dim3(DEMB / 32, DEMB / 32),   tb, 0, stream>>>(Wq, WqT, DEMB,   DEMB);
  transpose_convert_kernel<<<dim3(DEMB / 32, DCROSS / 32), tb, 0, stream>>>(Wk, WkT, DCROSS, DEMB);
  transpose_convert_kernel<<<dim3(DEMB / 32, DCROSS / 32), tb, 0, stream>>>(Wv, WvT, DCROSS, DEMB);
  transpose_convert_kernel<<<dim3(DEMB / 32, DEMB / 32),   tb, 0, stream>>>(Wo, WoT, DEMB,   DEMB);

  // Q projection: MT=2 (32x64 wave tile, 128x64 block tile), f32 A on the fly.
  gemm_bt_kernel<2, float, half_t><<<dim3(DEMB / 64, (B_ * LQ) / 128), 128, 0, stream>>>(
      x, WqT, bq, Qh, B_ * LQ, DEMB, DEMB);
  // K/V projections: M=1232 (77 tiles of 16) -> MT=1 with wave guard, grid.y=20.
  gemm_bt_kernel<1, float, half_t><<<dim3(DEMB / 64, (B_ * LKV + 63) / 64), 128, 0, stream>>>(
      y, WkT, bk, Kh, B_ * LKV, DEMB, DCROSS);
  gemm_bt_kernel<1, float, half_t><<<dim3(DEMB / 64, (B_ * LKV + 63) / 64), 128, 0, stream>>>(
      y, WvT, bv, Vh, B_ * LKV, DEMB, DCROSS);

  const int nkp = B_ * KVPAD * DEMB;
  pad_k_kernel<<<(nkp + 255) / 256, 256, 0, stream>>>(Kh, Khp);
  const int nvt = B_ * DEMB * KVP;
  build_vt_kernel<<<(nvt + 255) / 256, 256, 0, stream>>>(Vh, Vt);

  attention_kernel<<<(B_ * (LQ / 16)) / 8, 256, 0, stream>>>(Qh, Khp, Vt, Oh);

  // Output projection: f16 A, f32 out, MT=2.
  gemm_bt_kernel<2, half_t, float><<<dim3(DEMB / 64, (B_ * LQ) / 128), 128, 0, stream>>>(
      Oh, WoT, bo, out, B_ * LQ, DEMB, DEMB);
}
